// LinearSHTP_48567490183833
// MI455X (gfx1250) — matrix operand order repaired and checked by reading, not compiled, not run on hardware
//
#include <hip/hip_runtime.h>

// ---------------------------------------------------------------------------
// LinearSHTP for MI455X (gfx1250, wave32, WMMA).
//
// Roofline: 53 x (32768x64)@(64x64) GEMMs = 14.2 GFLOP; 285 MB HBM traffic
// -> ~12us memory floor @ 23.3 TB/s. bf16 WMMA (16x16x32, f32 acc) pushes
// compute far under the memory floor; fp32 WMMA (16x16x4) would be ~8x
// slower on the matrix pipe and compute-bound. So: fp32 everywhere except
// the channel-mixing GEMMs, which run v_wmma_f32_16x16x32_bf16.
// ---------------------------------------------------------------------------

typedef __attribute__((ext_vector_type(16))) __bf16 v16bf;
typedef __attribute__((ext_vector_type(8)))  float  v8f;

union Frag16 { v16bf v; uint4 q[2]; };

// ---------------- contribution tables (derived from _pair_specs) -----------
// 53 contributions grouped by output irrep iz. Each: output column j within
// irrep, global input column (0..16), weight-matrix index (0..32), sign*scale.
constexpr int   kCStart[6]  = {0, 3, 14, 24, 39, 53};
constexpr int   kLz[5]      = {0, 1, 1, 2, 2};
constexpr int   kOutFlat[5] = {0, 64, 256, 448, 768};   // flat offsets in 1088
constexpr int   kInFlat[5]  = {0, 64, 256, 448, 768};
constexpr int   kColBase[5] = {0, 1, 4, 7, 12};         // global column bases

constexpr int cOutJ[53] = {
  0,0,0,
  0,0,0,0, 1,1,1, 2,2,2,2,
  0,0,0,0, 1,1, 2,2,2,2,
  0,0, 1,1,1,1, 2,2,2, 3,3,3,3, 4,4,
  0,0, 1,1,1,1, 2,2, 3,3,3,3, 4,4
};
constexpr int cInCol[53] = {
  0,2,9,
  1,6,8,15, 0,2,9, 3,4,10,13,
  3,4,10,13, 5,14, 1,6,8,15,
  7,16, 1,6,8,15, 0,2,9, 3,4,10,13, 11,12,
  11,12, 3,4,10,13, 5,14, 1,6,8,15, 7,16
};
constexpr int cWm[53] = {
  0,1,2,
  5,6,8,9, 3,4,7, 5,6,8,9,
  10,12,13,15, 11,14, 10,12,13,15,
  22,24, 18,19,21,23, 16,17,20, 18,19,21,23, 22,24,
  29,32, 25,27,28,31, 26,30, 25,27,28,31, 29,32
};
#define S0 0.07216878364870323f   /* 1/sqrt(192) */
#define S1 0.05590169943749474f   /* 1/sqrt(320) */
#define S2 0.0625f                /* 1/sqrt(256) */
constexpr float cSgnScale[53] = {
   S0, S0, S0,
   S1,-S1, S1,-S1,  S1, S1, S1,  S1, S1, S1, S1,
  -S2, S2,-S2, S2,  S2, S2,  S2, S2, S2, S2,
   S1,-S1,  S1,-S1, S1,-S1,  S1, S1, S1,  S1, S1, S1, S1,  S1, S1,
  -S2, S2, -S2, S2,-S2, S2,  S2, S2,  S2, S2, S2, S2,  S2, S2
};

// ---------------- Wigner constants -----------------------------------------
#define SQ32 0.8660254037844386f
constexpr float kJ1[9]  = { 0,1,0,  1,0,0,  0,0,-1 };
constexpr float kDP21[9]= { 1,0,0,  0,0,-1, 0,1,0 };
constexpr float kJ2[25] = {
  0,0,0,-1,0,   0,1,0,0,0,   0,0,-0.5f,0,-SQ32,
 -1,0,0,0,0,   0,0,-SQ32,0,0.5f };
constexpr float kDP22[25] = {
  0,1,0,0,0,  -1,0,0,0,0,   0,0,-0.5f,0,SQ32,
  0,0,0,-1,0,  0,0,SQ32,0,0.5f };

// ---------------- helpers ---------------------------------------------------
__device__ __forceinline__ unsigned short f2bf(float f) {
  unsigned int b = __float_as_uint(f);
  b += 0x7FFFu + ((b >> 16) & 1u);            // round-to-nearest-even
  return (unsigned short)(b >> 16);
}

template<int N>
__device__ __forceinline__ void mmN(const float* A, const float* B, float* C) {
  #pragma unroll
  for (int i = 0; i < N; ++i) {
    #pragma unroll
    for (int j = 0; j < N; ++j) {
      float s = 0.f;
      #pragma unroll
      for (int k = 0; k < N; ++k) s += A[i*N+k] * B[k*N+j];
      C[i*N+j] = s;
    }
  }
}

template<int L>
__device__ __forceinline__ void build_dy(float th, float* M) {
  constexpr int N = 2*L+1;
  #pragma unroll
  for (int i = 0; i < N*N; ++i) M[i] = 0.f;
  #pragma unroll
  for (int i = 0; i < N; ++i) {
    float m = (float)(i - L);
    M[i*N + i]        += cosf(m * th);
    M[i*N + (N-1-i)]  += -sinf(m * th);
  }
}

// M_in[l] = (gl ? D_PI2[l] : I) @ D(l, alpha, beta);  applied as c @ M_in.
__device__ void compute_rot(const float* __restrict__ dir, int b,
                            float* M1, float* M2) {
  float dx = dir[3*b+0], dy = dir[3*b+1], dz = dir[3*b+2];
  float n2 = dx*dx + dy*dy + dz*dz;
  float rn = rsqrtf(n2 > 0.f ? n2 : 1.f);
  dx *= rn; dy *= rn; dz *= rn;
  const bool gl = fabsf(dy) > 0.99f;
  if (gl) { float ny = dz, nz = -dy; dy = ny; dz = nz; }  // d @ D_PI2[1]
  float alpha = atan2f(dx, dz);
  float beta  = acosf(fminf(fmaxf(dy, -1.f), 1.f));
  float A[25], Bm[25], T1[25], T2[25], D[25];
  // l = 1
  build_dy<1>(alpha, A); build_dy<1>(beta, Bm);
  mmN<3>(A, kJ1, T1); mmN<3>(T1, Bm, T2); mmN<3>(T2, kJ1, D);
  if (gl) {
    mmN<3>(kDP21, D, M1);
  } else {
    #pragma unroll
    for (int q = 0; q < 9; ++q) M1[q] = D[q];
  }
  // l = 2
  build_dy<2>(alpha, A); build_dy<2>(beta, Bm);
  mmN<5>(A, kJ2, T1); mmN<5>(T1, Bm, T2); mmN<5>(T2, kJ2, D);
  if (gl) {
    mmN<5>(kDP22, D, M2);
  } else {
    #pragma unroll
    for (int q = 0; q < 25; ++q) M2[q] = D[q];
  }
}

// ---------------- stage 1: weights -> 53 signed/scaled bf16 matrices -------
// Stored transposed (v-major, u contiguous) so a WMMA B-fragment lane reads
// two runs of 8 bf16 (K = 32s+8h+0..7 and 32s+16+8h+0..7 for column N = 16t + lane%16).
__global__ __launch_bounds__(256)
void shtp_prep_w(const float* __restrict__ w, unsigned short* __restrict__ Wbf) {
  int g = blockIdx.x * 256 + threadIdx.x;
  if (g >= 53 * 4096) return;
  int c = g >> 12, r = g & 4095, v = r >> 6, u = r & 63;
  float val = w[(size_t)cWm[c] * 4096 + u * 64 + v] * cSgnScale[c];
  Wbf[(size_t)c * 4096 + (size_t)v * 64 + u] = f2bf(val);
}

// ---------------- stage 2: per-sample rotation + bf16 repack ---------------
// Ybf[b][col][u]: u contiguous -> A fragment = two b128 loads per k-step.
__global__ __launch_bounds__(256)
void shtp_prep_x(const float* __restrict__ x, const float* __restrict__ dir,
                 unsigned short* __restrict__ Ybf, float* __restrict__ Mws,
                 int B) {
  __shared__ float sM[4][36];
  const int sl = threadIdx.x >> 6;
  const int u  = threadIdx.x & 63;
  const int b  = blockIdx.x * 4 + sl;
  if (b < B && u == 0) {
    float M1[9], M2[25];
    compute_rot(dir, b, M1, M2);
    #pragma unroll
    for (int q = 0; q < 9;  ++q) { sM[sl][q]   = M1[q]; Mws[(size_t)b*40+q]   = M1[q]; }
    #pragma unroll
    for (int q = 0; q < 25; ++q) { sM[sl][9+q] = M2[q]; Mws[(size_t)b*40+9+q] = M2[q]; }
  }
  __syncthreads();
  if (b >= B) return;
  const float* xr = x + (size_t)b * 1088;
  unsigned short* yr = Ybf + (size_t)b * 1088;
  // irrep 0 (l=0): identity rotation
  yr[u] = f2bf(xr[u]);
  float cb[5];
  // irreps 1,2 (l=1)
  #pragma unroll
  for (int ir = 1; ir <= 2; ++ir) {
    const int fb = kInFlat[ir], cbase = kColBase[ir];
    #pragma unroll
    for (int j = 0; j < 3; ++j) cb[j] = xr[fb + u*3 + j];
    #pragma unroll
    for (int jn = 0; jn < 3; ++jn) {
      float a = 0.f;
      #pragma unroll
      for (int j = 0; j < 3; ++j) a += cb[j] * sM[sl][j*3 + jn];
      yr[(cbase + jn) * 64 + u] = f2bf(a);
    }
  }
  // irreps 3,4 (l=2)
  #pragma unroll
  for (int ir = 3; ir <= 4; ++ir) {
    const int fb = kInFlat[ir], cbase = kColBase[ir];
    #pragma unroll
    for (int j = 0; j < 5; ++j) cb[j] = xr[fb + u*5 + j];
    #pragma unroll
    for (int jn = 0; jn < 5; ++jn) {
      float a = 0.f;
      #pragma unroll
      for (int j = 0; j < 5; ++j) a += cb[j] * sM[sl][9 + j*5 + jn];
      yr[(cbase + jn) * 64 + u] = f2bf(a);
    }
  }
}

// ---------------- stage 3: WMMA GEMMs + fused output rotation --------------
// One wave per 16-row batch tile; templated on output irrep so contribution
// tables fold to compile-time constants (accumulators stay in VGPRs).
__device__ __forceinline__ v8f wmma_bf16(v16bf a, v16bf b, v8f c) {
  return __builtin_amdgcn_wmma_f32_16x16x32_bf16(false, a, false, b,
                                                 (short)0, c, false, false);
}

template<int IZ>
__global__ __launch_bounds__(32)
void shtp_gemm(const unsigned short* __restrict__ Ybf,
               const unsigned short* __restrict__ Wbf,
               const float* __restrict__ Mws,
               float* __restrict__ out, int B) {
  constexpr int LZ   = kLz[IZ];
  constexpr int NC   = 2*LZ + 1;
  constexpr int C0   = kCStart[IZ], C1 = kCStart[IZ+1];
  constexpr int FLAT = kOutFlat[IZ];
  constexpr int MOFF = (LZ == 2) ? 9 : 0;

  __shared__ float zs[16][64 * NC];

  const int lane = threadIdx.x;
  const int h = lane >> 4, n = lane & 15;
  const int bt = blockIdx.x * 16;
  if (bt >= B) return;

  v8f acc[NC][4];
  const v8f vzero = {0.f,0.f,0.f,0.f,0.f,0.f,0.f,0.f};
  #pragma unroll
  for (int j = 0; j < NC; ++j) {
    #pragma unroll
    for (int t = 0; t < 4; ++t) acc[j][t] = vzero;
  }

  const unsigned short* __restrict__ yrow = Ybf + (size_t)(bt + n) * 1088;

  #pragma unroll
  for (int c = C0; c < C1; ++c) {
    const int ic = cInCol[c];
    const int j  = cOutJ[c];
    // A fragments (16x32 bf16): lane holds K = 32s+8h+0..7 and 32s+16+8h+0..7
    Frag16 a0, a1;
    const unsigned short* pa = yrow + ic * 64;
    a0.q[0] = *(const uint4*)(pa + 8*h);
    a0.q[1] = *(const uint4*)(pa + 16 + 8*h);
    a1.q[0] = *(const uint4*)(pa + 32 + 8*h);
    a1.q[1] = *(const uint4*)(pa + 48 + 8*h);
    #pragma unroll
    for (int t = 0; t < 4; ++t) {
      // B fragments (32x16 bf16): lane holds K = 32s+8h+0..7 and 32s+16+8h+0..7, N = 16t+n
      const unsigned short* pb =
          Wbf + (size_t)c * 4096 + (size_t)(16*t + n) * 64 + 8*h;
      Frag16 b0, b1;
      b0.q[0] = *(const uint4*)(pb + 0);
      b0.q[1] = *(const uint4*)(pb + 16);
      b1.q[0] = *(const uint4*)(pb + 32);
      b1.q[1] = *(const uint4*)(pb + 48);
      acc[j][t] = wmma_bf16(a0.v, b0.v, acc[j][t]);
      acc[j][t] = wmma_bf16(a1.v, b1.v, acc[j][t]);
    }
  }

  // Epilogue: z_final[:, j] = sum_j' z_rot[:, j'] * M_in[j][j']  (M_out=M_in^T)
  // C/D layout: VGPR g -> batch row bt + 8h + g, channel v = 16t + n.
  #pragma unroll
  for (int g = 0; g < 8; ++g) {
    const int r = bt + 8*h + g;
    float M[NC*NC];
    if constexpr (LZ > 0) {
      const float* mp = Mws + (size_t)r * 40 + MOFF;
      #pragma unroll
      for (int q = 0; q < NC*NC; ++q) M[q] = mp[q];
    }
    #pragma unroll
    for (int t = 0; t < 4; ++t) {
      const int v = 16*t + n;
      #pragma unroll
      for (int j = 0; j < NC; ++j) {
        float val;
        if constexpr (LZ == 0) {
          val = acc[0][t][g];
        } else {
          val = 0.f;
          #pragma unroll
          for (int jp = 0; jp < NC; ++jp) val += acc[jp][t][g] * M[j*NC + jp];
        }
        zs[8*h + g][v*NC + j] = val;
      }
    }
  }

  // Coalesced fp32 store of this irrep's segment of each row.
  for (int rr = 0; rr < 16; ++rr) {
    float* orow = out + (size_t)(bt + rr) * 1088 + FLAT;
    for (int idx = lane; idx < 64 * NC; idx += 32) orow[idx] = zs[rr][idx];
  }
}

// ---------------- launcher --------------------------------------------------
extern "C" void kernel_launch(void* const* d_in, const int* in_sizes, int n_in,
                              void* d_out, int out_size, void* d_ws, size_t ws_size,
                              hipStream_t stream) {
  (void)n_in; (void)out_size; (void)ws_size;
  const float* x   = (const float*)d_in[0];
  const float* dir = (const float*)d_in[1];
  const float* w   = (const float*)d_in[2];
  float* out = (float*)d_out;
  const int B = in_sizes[0] / 1088;

  char* ws = (char*)d_ws;
  unsigned short* Wbf = (unsigned short*)ws;          // 53*4096*2  = 434176 B
  size_t off = (size_t)53 * 4096 * 2;
  float* Mws = (float*)(ws + off);                    // B*40*4
  off += (size_t)B * 40 * 4;
  unsigned short* Ybf = (unsigned short*)(ws + off);  // B*1088*2

  shtp_prep_w<<<(53 * 4096 + 255) / 256, 256, 0, stream>>>(w, Wbf);
  shtp_prep_x<<<(B + 3) / 4, 256, 0, stream>>>(x, dir, Ybf, Mws, B);

  const int gx = (B + 15) / 16;
  shtp_gemm<0><<<gx, 32, 0, stream>>>(Ybf, Wbf, Mws, out, B);
  shtp_gemm<1><<<gx, 32, 0, stream>>>(Ybf, Wbf, Mws, out, B);
  shtp_gemm<2><<<gx, 32, 0, stream>>>(Ybf, Wbf, Mws, out, B);
  shtp_gemm<3><<<gx, 32, 0, stream>>>(Ybf, Wbf, Mws, out, B);
  shtp_gemm<4><<<gx, 32, 0, stream>>>(Ybf, Wbf, Mws, out, B);
}
